// Streaming_85048942395816
// MI455X (gfx1250) — matrix operand order repaired and checked by reading, not compiled, not run on hardware
//
#include <hip/hip_runtime.h>
#include <hip/hip_bf16.h>
#include <stdint.h>

typedef __attribute__((ext_vector_type(16))) __bf16 v16bf;
typedef __attribute__((ext_vector_type(8)))  float  v8f;
typedef __attribute__((ext_vector_type(4)))  float  v4f;

#define DIMS   512
#define NQ     128
#define NCAND  262144
#define K_OUT  100

// ---- kernel A (GEMM) tiling ----
#define QT     64                     // queries per workgroup (2 query tiles total)
#define CHUNK  512                    // candidates per workgroup
#define WAVES  16
#define BLOCK  (WAVES * 32)           // 512 threads

// ---- kernel B (selection) ----
#define CH     16384                  // candidates per selection wave
#define NCH    (NCAND / CH)           // 16 partials per query

typedef unsigned long long u64;

// ---------- packed (score, index) key: order by score, ties -> smaller index ----------
__device__ __forceinline__ u64 packKey(float s, int idx) {
    unsigned u = __float_as_uint(s);
    u = (u & 0x80000000u) ? ~u : (u | 0x80000000u);   // order-preserving float->uint
    return ((u64)u << 32) | (unsigned)(~idx);
}
__device__ __forceinline__ float keyScore(u64 k) {
    unsigned e = (unsigned)(k >> 32);
    unsigned u = (e & 0x80000000u) ? (e ^ 0x80000000u) : ~e;
    return __uint_as_float(u);
}
__device__ __forceinline__ int keyIdx(u64 k) { return (int)(~(unsigned)k); }

// ---------- wave32 64-bit shuffles ----------
__device__ __forceinline__ u64 shflx64(u64 v, int m) {
    unsigned lo = (unsigned)__shfl_xor((int)(unsigned)v, m, 32);
    unsigned hi = (unsigned)__shfl_xor((int)(v >> 32), m, 32);
    return ((u64)hi << 32) | lo;
}
__device__ __forceinline__ u64 bcast64(u64 v, int srcLane) {
    unsigned lo = (unsigned)__shfl((int)(unsigned)v, srcLane, 32);
    unsigned hi = (unsigned)__shfl((int)(v >> 32), srcLane, 32);
    return ((u64)hi << 32) | lo;
}

// ---------- bitonic network over 128 elems, element e = lane*4 + r ----------
__device__ __forceinline__ void bstep2(u64 T[4], int size, int stride, int lane) {
    if (stride >= 4) {
        const int lm = stride >> 2;
        #pragma unroll
        for (int r = 0; r < 4; ++r) {
            u64 p = shflx64(T[r], lm);
            int e = lane * 4 + r;
            bool desc = ((e & size) == 0);
            bool low  = ((lane & lm) == 0);
            bool keepA = ((T[r] > p) == (low == desc));
            T[r] = keepA ? T[r] : p;
        }
    } else {
        const int rb = stride;   // 1 or 2
        #pragma unroll
        for (int r = 0; r < 4; ++r) {
            if ((r & rb) == 0) {
                int e = lane * 4 + r;
                bool desc = ((e & size) == 0);
                u64 a = T[r], b = T[r | rb];
                u64 mx = a > b ? a : b;
                u64 mn = a > b ? b : a;
                T[r]      = desc ? mx : mn;
                T[r | rb] = desc ? mn : mx;
            }
        }
    }
}
__device__ __forceinline__ void sort128d2(u64 T[4], int lane) {
    #pragma unroll
    for (int size = 2; size <= 128; size <<= 1)
        #pragma unroll
        for (int stride = size >> 1; stride >= 1; stride >>= 1)
            bstep2(T, size, stride, lane);
}
__device__ __forceinline__ void merge128d2(u64 T[4], int lane) {
    #pragma unroll
    for (int stride = 64; stride >= 1; stride >>= 1)
        bstep2(T, 128, stride, lane);
}
// merge sorted-desc tile T into sorted-desc running R (exact top-128 of union)
__device__ __forceinline__ void topkMerge2(u64 R[4], const u64 T[4], int lane) {
    #pragma unroll
    for (int r = 0; r < 4; ++r) {
        u64 tr = shflx64(T[3 - r], 31);   // reversed -> ascending
        R[r] = R[r] > tr ? R[r] : tr;
    }
    merge128d2(R, lane);
}

// =====================================================================
// Kernel A: bf16x3 split-precision WMMA GEMM -> full score matrix in ws.
// Each wave: 2 N-tiles x 4 M-tiles share one B conversion (12 WMMA/conv).
// Score stores are non-temporal (no reuse before kernel B, > L2 size).
// =====================================================================
__global__ __launch_bounds__(BLOCK) void gemm_scores(
    const float* __restrict__ qemb, const float* __restrict__ cand,
    float* __restrict__ gS)
{
    __shared__ __align__(16) __bf16 sAhi[QT][DIMS];   // 64 KB
    __shared__ __align__(16) __bf16 sAlo[QT][DIMS];   // 64 KB

    const int tid   = threadIdx.x;
    const int qbase = blockIdx.x * QT;    // x fast-varying: both q-tiles share chunk in L2
    const int cbase = blockIdx.y * CHUNK;

    for (int i = tid; i < QT * DIMS; i += BLOCK) {
        int qq = i >> 9, dd = i & (DIMS - 1);
        float f = qemb[(qbase + qq) * DIMS + dd];
        __bf16 h = (__bf16)f;
        sAhi[qq][dd] = h;
        sAlo[qq][dd] = (__bf16)(f - (float)h);
    }
    __syncthreads();

    const int wave = tid >> 5, lane = tid & 31;
    const int l16  = lane & 15;
    const int aoff = (lane < 16) ? 0 : 8;
    const int koff = (lane < 16) ? 0 : 16;
    const int m0   = (lane < 16) ? 0 : 8;

    const float* brow[2];
    brow[0] = cand + (size_t)(cbase + (wave * 2 + 0) * 16 + l16) * DIMS;
    brow[1] = cand + (size_t)(cbase + (wave * 2 + 1) * 16 + l16) * DIMS;

    v8f acc[2][4];
    #pragma unroll
    for (int nt = 0; nt < 2; ++nt)
        #pragma unroll
        for (int m = 0; m < 4; ++m)
            acc[nt][m] = (v8f){0.f,0.f,0.f,0.f,0.f,0.f,0.f,0.f};

    #pragma unroll 2
    for (int kk = 0; kk < 16; ++kk) {
        const int k0 = kk * 32;
        union { uint4 u[2]; v16bf v; } ahi[4], alo[4];
        #pragma unroll
        for (int m = 0; m < 4; ++m) {
            const int row = m * 16 + l16;
            ahi[m].u[0] = *(const uint4*)&sAhi[row][k0 + aoff];
            ahi[m].u[1] = *(const uint4*)&sAhi[row][k0 + 16 + aoff];
            alo[m].u[0] = *(const uint4*)&sAlo[row][k0 + aoff];
            alo[m].u[1] = *(const uint4*)&sAlo[row][k0 + 16 + aoff];
        }
        #pragma unroll
        for (int nt = 0; nt < 2; ++nt) {
            union { float f[16]; v4f q4[4]; } bfv;
            const v4f* bp = (const v4f*)(brow[nt] + k0 + aoff);
            bfv.q4[0] = bp[0]; bfv.q4[1] = bp[1]; bfv.q4[2] = bp[4]; bfv.q4[3] = bp[5];
            union { __bf16 h[16]; v16bf v; } bhi, blo;
            #pragma unroll
            for (int i = 0; i < 16; ++i) {
                __bf16 h = (__bf16)bfv.f[i];
                bhi.h[i] = h;
                blo.h[i] = (__bf16)(bfv.f[i] - (float)h);
            }
            #pragma unroll
            for (int m = 0; m < 4; ++m) {
                acc[nt][m] = __builtin_amdgcn_wmma_f32_16x16x32_bf16(false, ahi[m].v, false, bhi.v, (short)0, acc[nt][m], false, false);
                acc[nt][m] = __builtin_amdgcn_wmma_f32_16x16x32_bf16(false, ahi[m].v, false, blo.v, (short)0, acc[nt][m], false, false);
                acc[nt][m] = __builtin_amdgcn_wmma_f32_16x16x32_bf16(false, alo[m].v, false, bhi.v, (short)0, acc[nt][m], false, false);
            }
        }
    }

    #pragma unroll
    for (int nt = 0; nt < 2; ++nt) {
        const int col = cbase + (wave * 2 + nt) * 16 + l16;
        #pragma unroll
        for (int m = 0; m < 4; ++m)
            #pragma unroll
            for (int r = 0; r < 8; ++r)
                __builtin_nontemporal_store(acc[nt][m][r],
                    &gS[(size_t)(qbase + m * 16 + m0 + r) * NCAND + col]);
    }
}

// =====================================================================
// Kernel B: threshold-filtered streaming top-128 per (query, 16K chunk).
// Survivors compacted into an LDS buffer; bitonic sort+merge only when
// 128 accumulate (exact: tau-filter drops only keys < eventual 128th).
// Double-buffered NT loads + global_prefetch 8KB ahead to hide latency.
// =====================================================================
__global__ __launch_bounds__(256) void select_topk(
    const float* __restrict__ gS, u64* __restrict__ part)
{
    __shared__ u64 buf[8][256];          // 16 KB, one 256-entry buffer per wave
    const int lane = threadIdx.x & 31;
    const int wv   = threadIdx.x >> 5;
    const int gw   = blockIdx.x * 8 + wv;          // 0..2047
    const int q    = gw >> 4;                      // NCH = 16
    const int chunk = gw & (NCH - 1);
    const float* src = gS + (size_t)q * NCAND + (size_t)chunk * CH;
    const int idx0 = chunk * CH;
    u64* wb = buf[wv];

    u64 R[4] = {0, 0, 0, 0};
    u64 tau = 0;
    int count = 0;

    const int NT = CH / 128;                       // 128 tiles
    v4f f = __builtin_nontemporal_load((const v4f*)(src + lane * 4));
    for (int t = 0; t < NT; ++t) {
        // issue next tile's load + a deep prefetch before the filter work
        const int tn = (t + 1 < NT) ? t + 1 : t;
        v4f fn = __builtin_nontemporal_load((const v4f*)(src + tn * 128 + lane * 4));
        if (t + 16 < NT)
            __builtin_prefetch(src + (t + 16) * 128 + lane * 4, 0, 0);

        const int ib = idx0 + t * 128 + lane * 4;
        u64 key[4] = { packKey(f.x, ib + 0), packKey(f.y, ib + 1),
                       packKey(f.z, ib + 2), packKey(f.w, ib + 3) };
        #pragma unroll
        for (int r = 0; r < 4; ++r) {
            bool pred = key[r] > tau;
            u64 m = __ballot(pred);
            if (m) {                                    // wave-uniform
                int pos = count + (int)__popcll(m & ((1ull << lane) - 1ull));
                if (pred) wb[pos] = key[r];
                count += (int)__popcll(m);
            }
        }
        if (count >= 128) {                             // wave-uniform flush
            count -= 128;
            u64 T[4];
            #pragma unroll
            for (int r = 0; r < 4; ++r) T[r] = wb[count + lane * 4 + r];
            sort128d2(T, lane);
            topkMerge2(R, T, lane);
            tau = bcast64(R[3], 31);                    // element 127 = lane31,r3
        }
        f = fn;
    }
    // drain remaining (< 128) survivors
    {
        u64 T[4];
        #pragma unroll
        for (int r = 0; r < 4; ++r) {
            int p = lane * 4 + r;
            T[r] = (p < count) ? wb[p] : 0ull;
        }
        sort128d2(T, lane);
        topkMerge2(R, T, lane);
    }
    u64* dst = part + ((size_t)q * NCH + chunk) * 128;
    #pragma unroll
    for (int r = 0; r < 4; ++r) dst[lane * 4 + r] = R[r];
}

// =====================================================================
// Kernel C: one wave per query merges NCH sorted partials -> top-100.
// =====================================================================
__global__ __launch_bounds__(32) void merge_topk(
    const u64* __restrict__ part, float* __restrict__ outS, int* __restrict__ outI)
{
    const int q = blockIdx.x, lane = threadIdx.x;
    u64 R[4] = {0, 0, 0, 0};
    for (int c = 0; c < NCH; ++c) {
        const u64* src = part + ((size_t)q * NCH + c) * 128;
        u64 T[4];
        #pragma unroll
        for (int r = 0; r < 4; ++r) T[r] = src[lane * 4 + r];
        topkMerge2(R, T, lane);
    }
    #pragma unroll
    for (int r = 0; r < 4; ++r) {
        int e = lane * 4 + r;
        if (e < K_OUT) {
            outS[q * K_OUT + e] = keyScore(R[r]);
            outI[q * K_OUT + e] = keyIdx(R[r]);
        }
    }
}

extern "C" void kernel_launch(void* const* d_in, const int* in_sizes, int n_in,
                              void* d_out, int out_size, void* d_ws, size_t ws_size,
                              hipStream_t stream) {
    const float* qemb = (const float*)d_in[0];   // [128,512] f32
    const float* cand = (const float*)d_in[1];   // [262144,512] f32
    // d_in[2] is k (=100), baked in as K_OUT.

    float* gS  = (float*)d_ws;                                   // 128*262144 f32 = 134.2 MB
    u64*   part = (u64*)((float*)d_ws + (size_t)NQ * NCAND);     // 128*16*128 u64 = 2 MB
    float* outS = (float*)d_out;
    int*   outI = (int*)d_out + NQ * K_OUT;

    dim3 gA(NQ / QT, NCAND / CHUNK);             // (2, 512)
    gemm_scores<<<gA, BLOCK, 0, stream>>>(qemb, cand, gS);
    select_topk<<<(NQ * NCH) / 8, 256, 0, stream>>>(gS, part);
    merge_topk<<<NQ, 32, 0, stream>>>(part, outS, outI);
}
